// PaTHAttention_82360292868547
// MI455X (gfx1250) — matrix operand order repaired and checked by reading, not compiled, not run on hardware
//
#include <hip/hip_runtime.h>
#include <hip/hip_bf16.h>

typedef __attribute__((ext_vector_type(16))) __bf16 bf16x16;
typedef __attribute__((ext_vector_type(8)))  float  f32x8;

// ---------------------------------------------------------------------------
// WMMA helpers (CDNA5 wave32, v_wmma_f32_16x16x32_bf16)
// ---------------------------------------------------------------------------
__device__ __forceinline__ f32x8 wmma_bf16(bf16x16 a, bf16x16 b, f32x8 c) {
  return __builtin_amdgcn_wmma_f32_16x16x32_bf16(
      /*neg_a=*/false, a, /*neg_b=*/false, b,
      /*c_mod=*/(short)0, c, /*reuse_a=*/false, /*reuse_b=*/false);
}

// A fragment: 16x32 bf16.  lanes 0-15: K 0..15; lanes 16-31: K 16..31
__device__ __forceinline__ bf16x16 frag_a(const float* p, int rs, int cs,
                                          int m0, int k0, int lane,
                                          float sign, const float* kscale) {
  int m = m0 + (lane & 15);
  int koff = (lane >> 4) * 16;
  bf16x16 r;
#pragma unroll
  for (int e = 0; e < 16; ++e) {
    int k = k0 + koff + e;
    float v = p[(size_t)m * rs + (size_t)k * cs] * sign;
    if (kscale) v *= kscale[k];
    r[e] = (__bf16)v;
  }
  return r;
}

// B fragment: 32x16 bf16. lanes 0-15 hold K=0..15, lanes 16-31 hold K=16..31
__device__ __forceinline__ bf16x16 frag_b(const float* p, int rs, int cs,
                                          int k0, int n0, int lane) {
  int n = n0 + (lane & 15);
  int kb = k0 + (lane >> 4) * 16;
  bf16x16 r;
#pragma unroll
  for (int e = 0; e < 16; ++e)
    r[e] = (__bf16)p[(size_t)(kb + e) * rs + (size_t)n * cs];
  return r;
}

__device__ __forceinline__ f32x8 frag_c(const float* p, int rs, int cs,
                                        int m0, int n0, int lane) {
  int n = n0 + (lane & 15);
  int mb = m0 + (lane >> 4) * 8;
  f32x8 r;
#pragma unroll
  for (int e = 0; e < 8; ++e) r[e] = p[(size_t)(mb + e) * rs + (size_t)n * cs];
  return r;
}

__device__ __forceinline__ void frag_store(float* p, int rs, int cs,
                                           int m0, int n0, int lane, f32x8 acc) {
  int n = n0 + (lane & 15);
  int mb = m0 + (lane >> 4) * 8;
#pragma unroll
  for (int e = 0; e < 8; ++e) p[(size_t)(mb + e) * rs + (size_t)n * cs] = acc[e];
}

// D[64,64] = C + sign * (A(kscale) @ B), K steps of 32, 16 tiles over 8 waves.
__device__ __forceinline__ void mm64(float* D, int Drs, int Dcs,
                                     const float* A, int Ars, int Acs,
                                     const float* kscale, float asign,
                                     const float* B, int Brs, int Bcs,
                                     const float* C, int Crs, int Ccs,
                                     int K, int tid) {
  int lane = tid & 31, wave = tid >> 5;
#pragma unroll
  for (int t = 0; t < 2; ++t) {
    int tt = wave + t * 8;
    int m0 = (tt >> 2) * 16, n0 = (tt & 3) * 16;
    f32x8 acc = {0.f, 0.f, 0.f, 0.f, 0.f, 0.f, 0.f, 0.f};
    if (C) acc = frag_c(C, Crs, Ccs, m0, n0, lane);
    for (int k0 = 0; k0 < K; k0 += 32) {
      bf16x16 a = frag_a(A, Ars, Acs, m0, k0, lane, asign, kscale);
      bf16x16 b = frag_b(B, Brs, Bcs, k0, n0, lane);
      acc = wmma_bf16(a, b, acc);
    }
    frag_store(D, Drs, Dcs, m0, n0, lane, acc);
  }
}

// ---------------------------------------------------------------------------
// Generic GEMM: C[M,N] = A[M,K] @ Bt[N,K]^T (+bias), fp32 io, bf16 WMMA math.
// Block 256 = 8 waves, block tile 64x128; wave tile 32x32 (4 accumulators,
// A and B fragments each reused twice -> 4 WMMAs per k-step per wave).
// ---------------------------------------------------------------------------
__global__ __launch_bounds__(256) void gemm_xbt(
    const float* __restrict__ A, int lda,
    const float* __restrict__ Bt, int ldb,
    const float* __restrict__ bias,
    float* __restrict__ Cd, int ldc, int M, int N, int K) {
  int lane = threadIdx.x & 31, wave = threadIdx.x >> 5;
  int m0 = blockIdx.y * 64 + (wave >> 2) * 32;   // 2 row-waves
  int n0 = blockIdx.x * 128 + (wave & 3) * 32;   // 4 col-waves
  f32x8 acc00 = {0.f,0.f,0.f,0.f,0.f,0.f,0.f,0.f};
  f32x8 acc01 = acc00, acc10 = acc00, acc11 = acc00;

  int mA0 = m0 + (lane & 15);       mA0 = (mA0 < M) ? mA0 : 0;
  int mA1 = m0 + 16 + (lane & 15);  mA1 = (mA1 < M) ? mA1 : 0;
  int koff = (lane >> 4) * 16;
  int nB = n0 + (lane & 15);
  const float* ap0 = A  + (size_t)mA0 * lda;
  const float* ap1 = A  + (size_t)mA1 * lda;
  const float* bp0 = Bt + (size_t)((nB      < N) ? nB      : 0) * ldb;
  const float* bp1 = Bt + (size_t)((nB + 16 < N) ? nB + 16 : 0) * ldb;

#pragma unroll 2
  for (int k0 = 0; k0 < K; k0 += 32) {
    if (k0 + 64 < K) {              // warm next K tiles (global_prefetch_b8)
      __builtin_prefetch(ap0 + k0 + 64, 0, 1);
      __builtin_prefetch(bp0 + k0 + 64, 0, 1);
    }
    bf16x16 a0, a1, b0, b1;
    int kb = k0 + (lane >> 4) * 16;
#pragma unroll
    for (int e = 0; e < 16; ++e) {
      int ka = k0 + koff + e;
      a0[e] = (__bf16)ap0[ka];
      a1[e] = (__bf16)ap1[ka];
      b0[e] = (__bf16)bp0[kb + e];
      b1[e] = (__bf16)bp1[kb + e];
    }
    acc00 = wmma_bf16(a0, b0, acc00);
    acc01 = wmma_bf16(a0, b1, acc01);
    acc10 = wmma_bf16(a1, b0, acc10);
    acc11 = wmma_bf16(a1, b1, acc11);
  }

  int n = n0 + (lane & 15);
  int mb = m0 + (lane >> 4) * 8;
  float bi0 = (bias && n      < N) ? bias[n]      : 0.f;
  float bi1 = (bias && n + 16 < N) ? bias[n + 16] : 0.f;
#pragma unroll
  for (int e = 0; e < 8; ++e) {
    int ma = mb + e, mc = mb + 16 + e;
    if (ma < M) {
      if (n      < N) Cd[(size_t)ma * ldc + n]      = acc00[e] + bi0;
      if (n + 16 < N) Cd[(size_t)ma * ldc + n + 16] = acc01[e] + bi1;
    }
    if (mc < M) {
      if (n      < N) Cd[(size_t)mc * ldc + n]      = acc10[e] + bi0;
      if (n + 16 < N) Cd[(size_t)mc * ldc + n + 16] = acc11[e] + bi1;
    }
  }
}

// ---------------------------------------------------------------------------
// Elementwise / small kernels
// ---------------------------------------------------------------------------
__global__ void rmsnorm_rows(float* x, const float* wgt, int rows, int hd) {
  int r = blockIdx.x * blockDim.x + threadIdx.x;
  if (r >= rows) return;
  float* p = x + (size_t)r * hd;
  float s = 0.f;
  for (int i = 0; i < hd; ++i) s += p[i] * p[i];
  float inv = rsqrtf(s / (float)hd + 1e-6f);
  for (int i = 0; i < hd; ++i) p[i] = wgt[i] * p[i] * inv;
}

__global__ void l2norm_rows(float* x, int rows, int hd) {
  int r = blockIdx.x * blockDim.x + threadIdx.x;
  if (r >= rows) return;
  float* p = x + (size_t)r * hd;
  float s = 0.f;
  for (int i = 0; i < hd; ++i) s += p[i] * p[i];
  float inv = rsqrtf(s + 1e-12f);
  for (int i = 0; i < hd; ++i) p[i] = p[i] * inv;
}

__global__ void sigmoid2_inplace(float* x, int n) {
  int i = blockIdx.x * blockDim.x + threadIdx.x;
  if (i < n) x[i] = 2.f / (1.f + expf(-x[i]));
}

__global__ void logsig_cumsum(const float* graw, float* G, int T, int H) {
  int h = blockIdx.x * blockDim.x + threadIdx.x;
  if (h >= H) return;
  float acc = 0.f;
  for (int t = 0; t < T; ++t) {
    float xv = graw[(size_t)t * H + h];
    float ls = fminf(xv, 0.f) - log1pf(expf(-fabsf(xv)));
    acc += ls;
    G[(size_t)t * H + h] = acc;
  }
}

__global__ void dwconv4(const float* __restrict__ x, const float* __restrict__ cw,
                        float* __restrict__ y, int T, int C) {
  int idx = blockIdx.x * blockDim.x + threadIdx.x;
  if (idx >= T * C) return;
  int t = idx / C, c = idx % C;
  float acc = 0.f;
#pragma unroll
  for (int kk = 0; kk < 4; ++kk) {
    int tt = t - 3 + kk;
    if (tt >= 0) acc += cw[c * 4 + kk] * x[(size_t)tt * C + c];
  }
  y[idx] = acc;
}

// ---------------------------------------------------------------------------
// PaTH attention core: one block per head, sequential over 16 chunks of 64.
// All 64x64 intermediates in LDS; every matmul is WMMA; triangular solves are
// row-parallel back-substitution; flash softmax over past chunks.
// ---------------------------------------------------------------------------
__global__ __launch_bounds__(256) void path_attn(
    const float* __restrict__ q, const float* __restrict__ k,
    const float* __restrict__ v, const float* __restrict__ w,
    const float* __restrict__ beta, const float* __restrict__ G,
    float* __restrict__ Kcar, float* __restrict__ o, int T, int H) {
  extern __shared__ float smem[];
  float* sQ  = smem;               // Qb -> Qt (in place)
  float* sK  = smem + 4096;        // Kb
  float* sW  = smem + 8192;        // Wb
  float* sM  = smem + 12288;       // WW-derived M, later O accumulator
  float* sR  = smem + 16384;       // QW -> R, later S/P, later Dnew
  float* sRw = smem + 20480;       // WW strict-lower -> Rw
  float* sD  = smem + 24576;       // WK strict-lower, later KV scratch
  float* sA  = smem + 28672;       // QK -> Aw, later Vh
  float* bb   = smem + 32768;      // 64
  float* gq   = bb + 64;
  float* gk   = gq + 64;
  float* mrow = gk + 64;
  float* lrow = mrow + 64;

  const int h = blockIdx.x;
  const int tid = (int)threadIdx.x;
  const int ld = H * 64;
  const float scale = 0.125f;      // 64^-0.5
  float* Kc = Kcar + (size_t)h * T * 64;
  const int nch = T / 64;

  for (int ic = 0; ic < nch; ++ic) {
    const int t0 = ic * 64;
    // --- load chunk ---
    for (int idx = tid; idx < 4096; idx += 256) {
      int r = idx >> 6, c = idx & 63;
      size_t off = (size_t)(t0 + r) * ld + (size_t)h * 64 + c;
      sQ[idx] = q[off];
      sK[idx] = k[off];
      sW[idx] = w[off];
    }
    if (tid < 64) {
      bb[tid]   = beta[(size_t)(t0 + tid) * H + h];
      gq[tid]   = G[(size_t)(t0 + tid) * H + h];
      mrow[tid] = -1e30f;
      lrow[tid] = 0.f;
    }
    __syncthreads();

    // WW = Wb Wb^T -> sRw
    mm64(sRw,64,1, sW,64,1, nullptr,1.f, sW,1,64, nullptr,0,0, 64, tid);
    __syncthreads();
    // M = I + diag(b) tril(WW,-1) -> sM ; sRw := tril(WW,-1)
    for (int idx = tid; idx < 4096; idx += 256) {
      int s = idx >> 6, u = idx & 63;
      float ww = sRw[idx];
      sM[idx]  = (s == u) ? 1.f : ((s > u) ? bb[s] * ww : 0.f);
      sRw[idx] = (s > u) ? ww : 0.f;
    }
    __syncthreads();
    // QW = Qb Wb^T -> sR, then tril
    mm64(sR,64,1, sQ,64,1, nullptr,1.f, sW,1,64, nullptr,0,0, 64, tid);
    __syncthreads();
    for (int idx = tid; idx < 4096; idx += 256) {
      int i = idx >> 6, s = idx & 63;
      if (s > i) sR[idx] = 0.f;
    }
    __syncthreads();
    // Back-substitution: X M = A  =>  X[i,s] = A[i,s] - sum_{u>s} X[i,u] M[u,s]
    if (tid < 64) {
      int i = tid;
      for (int s = 62; s >= 0; --s) {
        float acc = sR[i * 64 + s];
        for (int u = s + 1; u < 64; ++u) acc -= sR[i * 64 + u] * sM[u * 64 + s];
        sR[i * 64 + s] = acc;
      }
    } else if (tid < 128) {
      int i = tid - 64;
      for (int s = 62; s >= 0; --s) {
        float acc = sRw[i * 64 + s];
        for (int u = s + 1; u < 64; ++u) acc -= sRw[i * 64 + u] * sM[u * 64 + s];
        sRw[i * 64 + s] = acc;
      }
    }
    __syncthreads();
    // WK = Wb Kb^T -> sD, keep strict lower
    mm64(sD,64,1, sW,64,1, nullptr,1.f, sK,1,64, nullptr,0,0, 64, tid);
    __syncthreads();
    for (int idx = tid; idx < 4096; idx += 256) {
      int s = idx >> 6, j = idx & 63;
      if (s <= j) sD[idx] = 0.f;
    }
    __syncthreads();
    // Aw = Qb Kb^T - (R b) sWK -> sA
    mm64(sA,64,1, sQ,64,1, nullptr,1.f, sK,1,64, nullptr,0,0, 64, tid);
    __syncthreads();
    mm64(sA,64,1, sR,64,1, bb,-1.f, sD,64,1, sA,64,1, 64, tid);
    __syncthreads();
    // Qt = Qb - (R b) Wb  (in place into sQ)
    mm64(sQ,64,1, sR,64,1, bb,-1.f, sW,64,1, sQ,64,1, 64, tid);
    __syncthreads();

    // --- flash softmax over chunks 0..ic ---
    float* sO = sM;   // reuse
    float* sS = sR;   // reuse
    for (int idx = tid; idx < 4096; idx += 256) sO[idx] = 0.f;
    __syncthreads();
    for (int jc = 0; jc <= ic; ++jc) {
      if (tid < 64)
        gk[tid] = (jc < ic) ? G[(size_t)(jc * 64 + tid) * H + h] : gq[tid];
      __syncthreads();
      if (jc < ic) {
        // S = Qt @ Kcar_jc^T
        mm64(sS,64,1, sQ,64,1, nullptr,1.f, Kc + (size_t)jc * 4096, 1, 64,
             nullptr,0,0, 64, tid);
        __syncthreads();
        for (int idx = tid; idx < 4096; idx += 256) {
          int i = idx >> 6, j = idx & 63;
          sS[idx] = scale * sS[idx] + gq[i] - gk[j];
        }
      } else {
        for (int idx = tid; idx < 4096; idx += 256) {
          int i = idx >> 6, j = idx & 63;
          sS[idx] = (j <= i) ? (scale * sA[idx] + gq[i] - gk[j]) : -1e30f;
        }
      }
      __syncthreads();
      if (tid < 64) {
        int i = tid;
        float mold = mrow[i], mx = mold;
        for (int j = 0; j < 64; ++j) mx = fmaxf(mx, sS[i * 64 + j]);
        float alpha = expf(mold - mx);
        float l = lrow[i] * alpha;
        for (int j = 0; j < 64; ++j) {
          float p = expf(sS[i * 64 + j] - mx);
          sS[i * 64 + j] = p;
          l += p;
        }
        for (int c = 0; c < 64; ++c) sO[i * 64 + c] *= alpha;
        mrow[i] = mx;
        lrow[i] = l;
      }
      __syncthreads();
      // O += P @ V_jc   (V rows from global, stride ld)
      mm64(sO,64,1, sS,64,1, nullptr,1.f,
           v + (size_t)(jc * 64) * ld + (size_t)h * 64, ld, 1,
           sO,64,1, 64, tid);
      __syncthreads();
    }
    // write O / l
    for (int idx = tid; idx < 4096; idx += 256) {
      int i = idx >> 6, c = idx & 63;
      o[(size_t)(t0 + i) * ld + (size_t)h * 64 + c] = sO[idx] / lrow[i];
    }
    __syncthreads();

    // --- carry update ---
    float* sVh = sA;  // reuse (Aw dead)
    // Vh = Wb - (Rw b) Wb
    mm64(sVh,64,1, sRw,64,1, bb,-1.f, sW,64,1, sW,64,1, 64, tid);
    __syncthreads();
    // Dnew = sWK - (Rw b) sWK  -> sR
    mm64(sR,64,1, sRw,64,1, bb,-1.f, sD,64,1, sD,64,1, 64, tid);
    __syncthreads();
    // Khat = Kb - (Dnew^T diag(b)) Wb  -> Kcar rows t0..t0+63
    mm64(Kc + (size_t)t0 * 64, 64,1, sR,1,64, bb,-1.f, sW,64,1, sK,64,1, 64, tid);
    __syncthreads();
    // previous rows: Knew = Kc - (Kc Vh^T diag(b)) Wb
    for (int pc = 0; pc < ic; ++pc) {
      float* kcp = Kc + (size_t)pc * 4096;
      mm64(sD,64,1, kcp,64,1, nullptr,1.f, sVh,1,64, nullptr,0,0, 64, tid);
      __syncthreads();
      mm64(kcp,64,1, sD,64,1, bb,-1.f, sW,64,1, kcp,64,1, 64, tid);
      __syncthreads();
    }
    __syncthreads();
  }
}

// ---------------------------------------------------------------------------
// Launch
// ---------------------------------------------------------------------------
extern "C" void kernel_launch(void* const* d_in, const int* in_sizes, int n_in,
                              void* d_out, int out_size, void* d_ws, size_t ws_size,
                              hipStream_t stream) {
  (void)in_sizes; (void)n_in; (void)out_size; (void)ws_size;
  const float* x      = (const float*)d_in[0];
  const float* Wq     = (const float*)d_in[1];
  const float* Wk     = (const float*)d_in[2];
  const float* Wv     = (const float*)d_in[3];
  const float* W1     = (const float*)d_in[4];
  const float* W2     = (const float*)d_in[5];
  const float* qnw    = (const float*)d_in[6];
  const float* knw    = (const float*)d_in[7];
  const float* convw  = (const float*)d_in[8];
  const float* btw    = (const float*)d_in[9];
  const float* btbias = (const float*)d_in[10];
  const float* gw     = (const float*)d_in[11];
  const float* gbias  = (const float*)d_in[12];
  const float* Wo     = (const float*)d_in[13];
  float* out = (float*)d_out;

  const int T = 1024, D = 2048, H = 32, hd = 64, rank = 32;
  float* ws = (float*)d_ws;
  const size_t TD = (size_t)T * D;
  float* qb   = ws + 0 * TD;
  float* kb   = ws + 1 * TD;
  float* vb   = ws + 2 * TD;
  float* wraw = ws + 3 * TD;
  float* wbuf = ws + 4 * TD;
  float* Kcar = ws + 5 * TD;          // H*T*64 == TD floats
  float* ob   = ws + 6 * TD;
  float* xw1  = ws + 7 * TD;          // T*rank
  float* btb  = xw1 + (size_t)T * rank;
  float* graw = btb + (size_t)T * H;
  float* Gb   = graw + (size_t)T * H;

  dim3 blk(256);
  dim3 gfull(D / 128, T / 64);        // 64x128 block tile
  dim3 gnarrow(1, T / 64);

  gemm_xbt<<<gfull, blk, 0, stream>>>(x, D, Wq, D, nullptr, qb, D, T, D, D);
  gemm_xbt<<<gfull, blk, 0, stream>>>(x, D, Wk, D, nullptr, kb, D, T, D, D);
  gemm_xbt<<<gfull, blk, 0, stream>>>(x, D, Wv, D, nullptr, vb, D, T, D, D);
  gemm_xbt<<<gnarrow, blk, 0, stream>>>(x, D, W1, D, nullptr, xw1, rank, T, rank, D);
  gemm_xbt<<<gfull, blk, 0, stream>>>(xw1, rank, W2, rank, nullptr, wraw, D, T, D, rank);
  gemm_xbt<<<gnarrow, blk, 0, stream>>>(x, D, btw, D, btbias, btb, H, T, H, D);
  gemm_xbt<<<gnarrow, blk, 0, stream>>>(x, D, gw, D, gbias, graw, H, T, H, D);

  rmsnorm_rows<<<(T * H + 255) / 256, 256, 0, stream>>>(qb, qnw, T * H, hd);
  rmsnorm_rows<<<(T * H + 255) / 256, 256, 0, stream>>>(kb, knw, T * H, hd);
  sigmoid2_inplace<<<(T * H + 255) / 256, 256, 0, stream>>>(btb, T * H);
  logsig_cumsum<<<1, 32, 0, stream>>>(graw, Gb, T, H);
  dwconv4<<<(T * D + 255) / 256, 256, 0, stream>>>(wraw, convw, wbuf, T, D);
  l2norm_rows<<<(T * H + 255) / 256, 256, 0, stream>>>(wbuf, T * H, hd);

  size_t shmem = (size_t)(8 * 4096 + 5 * 64) * sizeof(float);  // ~132 KB of 320 KB LDS
  path_attn<<<dim3(H), dim3(256), shmem, stream>>>(qb, kb, vb, wbuf, btb, Gb,
                                                   Kcar, ob, T, H);

  gemm_xbt<<<gfull, blk, 0, stream>>>(ob, D, Wo, D, nullptr, out, D, T, D, D);
}
